// ClassifierLSTM_62775241999107
// MI455X (gfx1250) — matrix operand order repaired and checked by reading, not compiled, not run on hardware
//
#include <hip/hip_runtime.h>
#include <hip/hip_bf16.h>

typedef __attribute__((ext_vector_type(16))) __bf16 v16bf;
typedef __attribute__((ext_vector_type(8)))  __bf16 v8bf;
typedef __attribute__((ext_vector_type(8)))  float  v8f;

#define HID   1024
#define EMB   512
#define GATES 4096
#define BATCH 64
#define TLEN  512

// ---------------- WMMA fragment loaders (16x16x32 bf16) ----------------
// A-matrix 16x32 (MxK), row-major source. Lane half (lane>>4):
//   elements 0..15 <-> K = k0 + half*16 .. +15 (contiguous, matches B)
__device__ __forceinline__ v16bf load_a_frag(const __bf16* A, int lda, int row0, int k0, int lane) {
    int m  = lane & 15;
    int hf = lane >> 4;
    const __bf16* p = A + (size_t)(row0 + m) * lda + k0 + hf * 16;
    v8bf lo = *(const v8bf*)p;
    v8bf hi = *(const v8bf*)(p + 8);
    v16bf r;
#pragma unroll
    for (int j = 0; j < 8; ++j) { r[j] = lo[j]; r[8 + j] = hi[j]; }
    return r;
}

// B-matrix 32x16 (KxN), loaded from an N-major (pre-transposed) weight BT[n][k]:
//   lanes 0-15 hold K = k0..k0+15, lanes 16-31 hold K = k0+16..k0+31 (contiguous)
__device__ __forceinline__ v16bf load_b_frag(const __bf16* BT, int ldb, int n0, int k0, int lane) {
    int n  = lane & 15;
    int hf = lane >> 4;
    const __bf16* p = BT + (size_t)(n0 + n) * ldb + k0 + hf * 16;
    v8bf lo = *(const v8bf*)p;
    v8bf hi = *(const v8bf*)(p + 8);
    v16bf r;
#pragma unroll
    for (int j = 0; j < 8; ++j) { r[j] = lo[j]; r[8 + j] = hi[j]; }
    return r;
}

__device__ __forceinline__ float sigmoidf_(float x) {
    return 1.0f / (1.0f + __expf(-x));
}

// ---------------- prep kernels ----------------
__global__ void embed_to_bf16_kernel(const int* __restrict__ inputs,
                                     const float* __restrict__ emb,
                                     __bf16* __restrict__ x) {
    int row = blockIdx.x;                 // b*T + t
    int tok = inputs[row];
    const float* e = emb + (size_t)tok * EMB;
    for (int j = threadIdx.x; j < EMB; j += blockDim.x)
        x[(size_t)row * EMB + j] = (__bf16)e[j];
}

// src [K, 4096] f32 row-major -> dst [4096, K] bf16 (N-major for B fragments)
__global__ void transpose_to_bf16_kernel(const float* __restrict__ src,
                                         __bf16* __restrict__ dst, int K) {
    int n = blockIdx.x;                   // 0..4095
    for (int k = threadIdx.x; k < K; k += blockDim.x)
        dst[(size_t)n * K + k] = (__bf16)src[(size_t)k * GATES + n];
}

__global__ void lengths_kernel(const int* __restrict__ inputs, int* __restrict__ idxv) {
    int b = threadIdx.x;
    if (b < BATCH) {
        int len = 0;
        for (int t = 0; t < TLEN; ++t) len += (inputs[b * TLEN + t] != 0);
        int idx = len - 1;
        idxv[b] = idx < 0 ? 0 : idx;
    }
}

// ---------------- big parallel GEMM: xz = X(bf16) @ KT^T + bias -> f32 ----------------
// X: [32768, K] bf16 row-major.  KT: [4096, K] bf16 N-major.  C: [32768, 4096] f32.
// One wave -> 16 rows x 64 cols (4 WMMA accumulators). 8 waves/block share the M-tile.
__global__ __launch_bounds__(256)
void gemm_xk_kernel(const __bf16* __restrict__ A, const __bf16* __restrict__ BT,
                    const float* __restrict__ bias, float* __restrict__ C, int K) {
    int lane = threadIdx.x & 31;
    int wave = blockIdx.x * (blockDim.x >> 5) + (threadIdx.x >> 5);
    int mt = wave >> 6;          // 0..2047  (16-row tile)
    int ng = wave & 63;          // 0..63    (64-col group)

    v8f acc0 = {}, acc1 = {}, acc2 = {}, acc3 = {};
    for (int kk = 0; kk < K; kk += 32) {
        v16bf a = load_a_frag(A, K, mt * 16, kk, lane);
        v16bf b0 = load_b_frag(BT, K, ng * 64 +  0, kk, lane);
        acc0 = __builtin_amdgcn_wmma_f32_16x16x32_bf16(false, a, false, b0, (short)0, acc0, false, false);
        v16bf b1 = load_b_frag(BT, K, ng * 64 + 16, kk, lane);
        acc1 = __builtin_amdgcn_wmma_f32_16x16x32_bf16(false, a, false, b1, (short)0, acc1, false, false);
        v16bf b2 = load_b_frag(BT, K, ng * 64 + 32, kk, lane);
        acc2 = __builtin_amdgcn_wmma_f32_16x16x32_bf16(false, a, false, b2, (short)0, acc2, false, false);
        v16bf b3 = load_b_frag(BT, K, ng * 64 + 48, kk, lane);
        acc3 = __builtin_amdgcn_wmma_f32_16x16x32_bf16(false, a, false, b3, (short)0, acc3, false, false);
    }

    int hf = lane >> 4;
    int nl = lane & 15;
    v8f accs[4] = {acc0, acc1, acc2, acc3};
#pragma unroll
    for (int j = 0; j < 4; ++j) {
        int col = ng * 64 + j * 16 + nl;
        float bv = bias[col];
#pragma unroll
        for (int r = 0; r < 8; ++r) {
            int row = mt * 16 + r + 8 * hf;
            C[(size_t)row * GATES + col] = accs[j][r] + bv;
        }
    }
}

// ---------------- fused recurrent step: z = xz[t] + h@R ; gates ; mask-carry ----------------
// One wave computes the SAME 16x16 (batch x hidden) block for all four gates,
// so i/f/g/o for a (b,hu) pair live in the same lane. 256 waves total.
__global__ __launch_bounds__(256)
void lstm_step_kernel(const __bf16* __restrict__ h_in,      // [64,1024] bf16 (read by all waves)
                      const __bf16* __restrict__ rT,        // [4096,1024] bf16 N-major
                      const float* __restrict__ xz,         // [B*T, 4096] f32
                      const int* __restrict__ inputs,       // [B, T]
                      const int* __restrict__ idxv,         // [B]
                      float* __restrict__ h_state,          // [64,1024] f32 (owner-lane private)
                      float* __restrict__ c_state,          // [64,1024] f32
                      __bf16* __restrict__ h_out,           // [64,1024] bf16 (next step's A)
                      __bf16* __restrict__ h_seq,           // [B*T,1024] bf16 (layer0 only)
                      float* __restrict__ final_h,          // [64,1024] f32 (layer1 only)
                      int t, int layer) {
    int lane = threadIdx.x & 31;
    int wave = blockIdx.x * (blockDim.x >> 5) + (threadIdx.x >> 5);
    int mt = wave & 3;           // batch tile (4 x 16 = 64)
    int nt = wave >> 2;          // hidden tile (64 x 16 = 1024)

    v8f zi = {}, zf = {}, zg = {}, zo = {};
    for (int kk = 0; kk < HID; kk += 32) {
        v16bf a = load_a_frag(h_in, HID, mt * 16, kk, lane);
        v16bf bi = load_b_frag(rT, HID, 0 * HID + nt * 16, kk, lane);
        zi = __builtin_amdgcn_wmma_f32_16x16x32_bf16(false, a, false, bi, (short)0, zi, false, false);
        v16bf bf_ = load_b_frag(rT, HID, 1 * HID + nt * 16, kk, lane);
        zf = __builtin_amdgcn_wmma_f32_16x16x32_bf16(false, a, false, bf_, (short)0, zf, false, false);
        v16bf bg = load_b_frag(rT, HID, 2 * HID + nt * 16, kk, lane);
        zg = __builtin_amdgcn_wmma_f32_16x16x32_bf16(false, a, false, bg, (short)0, zg, false, false);
        v16bf bo = load_b_frag(rT, HID, 3 * HID + nt * 16, kk, lane);
        zo = __builtin_amdgcn_wmma_f32_16x16x32_bf16(false, a, false, bo, (short)0, zo, false, false);
    }

    int hf = lane >> 4;
    int hu = nt * 16 + (lane & 15);
#pragma unroll
    for (int r = 0; r < 8; ++r) {
        int b = mt * 16 + r + 8 * hf;
        size_t xrow = ((size_t)b * TLEN + t) * GATES;
        float vi = sigmoidf_(zi[r] + xz[xrow + 0 * HID + hu]);
        float vf = sigmoidf_(zf[r] + xz[xrow + 1 * HID + hu]);
        float vg = tanhf    (zg[r] + xz[xrow + 2 * HID + hu]);
        float vo = sigmoidf_(zo[r] + xz[xrow + 3 * HID + hu]);

        size_t si = (size_t)b * HID + hu;
        float c_old = c_state[si];
        float h_old = h_state[si];
        float c_new = vf * c_old + vi * vg;
        float h_new = vo * tanhf(c_new);

        bool valid = inputs[b * TLEN + t] != 0;
        float c2 = valid ? c_new : c_old;
        float h2 = valid ? h_new : h_old;

        c_state[si] = c2;
        h_state[si] = h2;
        h_out[si]   = (__bf16)h2;
        if (layer == 0) {
            h_seq[((size_t)b * TLEN + t) * HID + hu] = (__bf16)h2;
        } else if (t == idxv[b]) {
            final_h[si] = h2;
        }
    }
}

// ---------------- tiny output head: logits = final_h @ Wout + bout ----------------
__global__ void logits_kernel(const float* __restrict__ final_h,
                              const float* __restrict__ Wout,
                              const float* __restrict__ bout,
                              float* __restrict__ out) {
    int i = threadIdx.x;
    if (i < BATCH * 3) {
        int b = i / 3, o = i % 3;
        float s = bout[o];
        const float* h = final_h + (size_t)b * HID;
        for (int k = 0; k < HID; ++k) s += h[k] * Wout[k * 3 + o];
        out[i] = s;
    }
}

extern "C" void kernel_launch(void* const* d_in, const int* in_sizes, int n_in,
                              void* d_out, int out_size, void* d_ws, size_t ws_size,
                              hipStream_t stream) {
    const int*   inputs = (const int*)  d_in[0];
    const float* emb    = (const float*)d_in[1];
    const float* k0     = (const float*)d_in[2];
    const float* r0     = (const float*)d_in[3];
    const float* b0     = (const float*)d_in[4];
    const float* k1     = (const float*)d_in[5];
    const float* r1     = (const float*)d_in[6];
    const float* b1     = (const float*)d_in[7];
    const float* Wout   = (const float*)d_in[8];
    const float* bout   = (const float*)d_in[9];
    float* out = (float*)d_out;

    // ---- workspace carve (256B aligned) ----
    size_t off = 0;
    char* base = (char*)d_ws;
    auto carve = [&](size_t bytes) -> void* {
        void* p = base + off;
        off += (bytes + 255) & ~(size_t)255;
        return p;
    };
    __bf16* x_bf   = (__bf16*)carve((size_t)BATCH * TLEN * EMB * 2);   //  32 MB
    __bf16* kT0    = (__bf16*)carve((size_t)GATES * EMB * 2);          //   4 MB
    __bf16* rT0    = (__bf16*)carve((size_t)GATES * HID * 2);          //   8 MB
    __bf16* kT1    = (__bf16*)carve((size_t)GATES * HID * 2);          //   8 MB
    __bf16* rT1    = (__bf16*)carve((size_t)GATES * HID * 2);          //   8 MB
    float*  xz     = (float*) carve((size_t)BATCH * TLEN * GATES * 4); // 512 MB (reused by both layers)
    __bf16* h_seq  = (__bf16*)carve((size_t)BATCH * TLEN * HID * 2);   //  64 MB
    float*  h_st   = (float*) carve((size_t)BATCH * HID * 4);
    float*  c_st   = (float*) carve((size_t)BATCH * HID * 4);
    __bf16* hbuf0  = (__bf16*)carve((size_t)BATCH * HID * 2);
    __bf16* hbuf1  = (__bf16*)carve((size_t)BATCH * HID * 2);
    float*  finalh = (float*) carve((size_t)BATCH * HID * 4);
    int*    idxv   = (int*)   carve(256);

    // ---- prep: embed lookup + weight transpose/convert + lengths ----
    embed_to_bf16_kernel<<<BATCH * TLEN, 128, 0, stream>>>(inputs, emb, x_bf);
    transpose_to_bf16_kernel<<<GATES, 256, 0, stream>>>(k0, kT0, EMB);
    transpose_to_bf16_kernel<<<GATES, 256, 0, stream>>>(r0, rT0, HID);
    transpose_to_bf16_kernel<<<GATES, 256, 0, stream>>>(k1, kT1, HID);
    transpose_to_bf16_kernel<<<GATES, 256, 0, stream>>>(r1, rT1, HID);
    lengths_kernel<<<1, 64, 0, stream>>>(inputs, idxv);

    const int gemmBlocks = (2048 * 64) / 8;  // 2048 M-tiles x 64 col-groups, 8 waves/block

    // ================= layer 0 =================
    hipMemsetAsync(h_st,  0, (size_t)BATCH * HID * 4, stream);
    hipMemsetAsync(c_st,  0, (size_t)BATCH * HID * 4, stream);
    hipMemsetAsync(hbuf0, 0, (size_t)BATCH * HID * 2, stream);
    hipMemsetAsync(hbuf1, 0, (size_t)BATCH * HID * 2, stream);
    hipMemsetAsync(finalh,0, (size_t)BATCH * HID * 4, stream);

    gemm_xk_kernel<<<gemmBlocks, 256, 0, stream>>>(x_bf, kT0, b0, xz, EMB);

    for (int t = 0; t < TLEN; ++t) {
        __bf16* hin  = (t & 1) ? hbuf1 : hbuf0;
        __bf16* hout = (t & 1) ? hbuf0 : hbuf1;
        lstm_step_kernel<<<32, 256, 0, stream>>>(hin, rT0, xz, inputs, idxv,
                                                 h_st, c_st, hout, h_seq, finalh,
                                                 t, /*layer=*/0);
    }

    // ================= layer 1 =================
    hipMemsetAsync(h_st,  0, (size_t)BATCH * HID * 4, stream);
    hipMemsetAsync(c_st,  0, (size_t)BATCH * HID * 4, stream);
    hipMemsetAsync(hbuf0, 0, (size_t)BATCH * HID * 2, stream);
    hipMemsetAsync(hbuf1, 0, (size_t)BATCH * HID * 2, stream);

    gemm_xk_kernel<<<gemmBlocks, 256, 0, stream>>>(h_seq, kT1, b1, xz, HID);

    for (int t = 0; t < TLEN; ++t) {
        __bf16* hin  = (t & 1) ? hbuf1 : hbuf0;
        __bf16* hout = (t & 1) ? hbuf0 : hbuf1;
        lstm_step_kernel<<<32, 256, 0, stream>>>(hin, rT1, xz, inputs, idxv,
                                                 h_st, c_st, hout, h_seq, finalh,
                                                 t, /*layer=*/1);
    }

    // ================= head =================
    logits_kernel<<<1, 256, 0, stream>>>(finalh, Wout, bout, out);
}
